// AIA_56049323213172
// MI455X (gfx1250) — matrix operand order repaired and checked by reading, not compiled, not run on hardware
//
#include <hip/hip_runtime.h>
#include <hip/hip_bf16.h>

// ---------------------------------------------------------------------------
// MI455X (gfx1250) implementation.
// All matrix work uses V_WMMA_F32_16X16X32_BF16 (bf16 in, f32 accumulate).
// Heavy GEMMs use 2x2 register blocking (32x32 output per wave) to halve
// L2 traffic vs a single 16x16 tile.
// ---------------------------------------------------------------------------

#define DEVINL __device__ __forceinline__

typedef __attribute__((ext_vector_type(16))) __bf16 v16bf;
typedef __attribute__((ext_vector_type(8)))  float  v8f;
typedef __attribute__((ext_vector_type(4)))  float  vf4;

static constexpr int C   = 512;
static constexpr int HW  = 4096;   // 64*64
static constexpr int C8  = 64;
static constexpr int PO  = 1024;   // 32*32 conv output pixels

// Workspace layout (bytes). Total ~64.1 MB.
static constexpr size_t OFF_XB  = 0;                          // x   (C,HW)  bf16
static constexpr size_t OFF_XT  = OFF_XB  + (size_t)C*HW*2;   // xT  (HW,C)  bf16
static constexpr size_t OFF_WB  = OFF_XT  + (size_t)HW*C*2;   // wpos (64,512) bf16
static constexpr size_t OFF_W3B = OFF_WB  + (size_t)C8*C*2;   // w3 repack (9,Co,Ci) bf16
static constexpr size_t OFF_YT  = OFF_W3B + (size_t)9*C*C*2;  // yT (4096,64) bf16
static constexpr size_t OFF_M   = OFF_YT  + (size_t)HW*C8*2;  // row max  f32
static constexpr size_t OFF_Z   = OFF_M   + (size_t)HW*4;     // 1/rowsum f32
static constexpr size_t OFF_X4T = OFF_Z   + (size_t)HW*4;     // x4^T (4096,4096) bf16
static constexpr size_t OFF_X6  = OFF_X4T + (size_t)HW*HW*2;  // x6 (C,HW) f32
static constexpr size_t OFF_X11 = OFF_X6  + (size_t)C*HW*4;   // x1_1^T (HW,C) bf16
static constexpr size_t OFF_X12 = OFF_X11 + (size_t)HW*C*2;   // x1_2 (C,PO) f32
static constexpr size_t OFF_C3  = OFF_X12 + (size_t)C*PO*4;   // c3   (C,PO) f32

union BfPack { v16bf v; vf4 q[2]; };

// A operand (16x32 bf16): lane holds row M=lane&15; 16 consecutive K at
// 16*(lane>>4).
DEVINL v16bf load_a_tile(const __bf16* t, int ld, int lane) {
  const __bf16* p = t + (size_t)(lane & 15) * ld + ((lane >> 4) << 4);
  BfPack u;
  u.q[0] = *(const vf4*)(p);
  u.q[1] = *(const vf4*)(p + 8);
  return u.v;
}
// B operand (32x16 bf16), stored N-major (row n contiguous over k):
// lane holds column N=lane&15, 16 consecutive K at 16*(lane>>4).
DEVINL v16bf load_b_tile(const __bf16* t, int ld, int lane) {
  const __bf16* p = t + (size_t)(lane & 15) * ld + ((lane >> 4) << 4);
  BfPack u;
  u.q[0] = *(const vf4*)(p);
  u.q[1] = *(const vf4*)(p + 8);
  return u.v;
}

DEVINL v8f wmma_bf16(v16bf a, v16bf b, v8f c) {
  return __builtin_amdgcn_wmma_f32_16x16x32_bf16(
      false, a, false, b, (short)0, c, false, false);
}

// ---------------------------------------------------------------------------
// Packing kernels
// ---------------------------------------------------------------------------
__global__ void k_pack_x(const float* __restrict__ x, __bf16* __restrict__ xb,
                         __bf16* __restrict__ xT) {
  int idx = blockIdx.x * 256 + threadIdx.x;
  if (idx >= C * HW) return;
  float v = x[idx];
  int c = idx >> 12, k = idx & (HW - 1);
  __bf16 b = (__bf16)v;
  xb[idx] = b;
  xT[(size_t)k * C + c] = b;
}

__global__ void k_pack_wpos(const float* __restrict__ w, __bf16* __restrict__ wb) {
  int idx = blockIdx.x * 256 + threadIdx.x;
  if (idx < C8 * C) wb[idx] = (__bf16)w[idx];
}

// w3 given as [co][ci][kh][kw] -> w3b[kh*3+kw][co][ci]
__global__ void k_pack_w3(const float* __restrict__ w3, __bf16* __restrict__ w3b) {
  int idx = blockIdx.x * 256 + threadIdx.x;
  if (idx >= C * C * 9) return;
  int kk = idx % 9;
  int tmp = idx / 9;
  int ci = tmp & (C - 1);
  int co = tmp >> 9;
  w3b[(size_t)kk * C * C + (size_t)co * C + ci] = (__bf16)w3[idx];
}

// ---------------------------------------------------------------------------
// conv1x1: y = w_pos (64x512) @ x (512x4096) + b_pos,
// stored transposed+spatially-permuted: yT[i][c8], i = w*64+h.
// ---------------------------------------------------------------------------
__global__ void __launch_bounds__(32)
k_conv1x1(const __bf16* __restrict__ wb, const __bf16* __restrict__ xT,
          const float* __restrict__ bpos, __bf16* __restrict__ yT) {
  int lane = threadIdx.x;
  int n0 = blockIdx.x << 4;   // column (natural spatial k)
  int m0 = blockIdx.y << 4;   // c8 row
  v8f acc = {};
  for (int k0 = 0; k0 < C; k0 += 32) {
    v16bf a = load_a_tile(wb + (size_t)m0 * C + k0, C, lane);
    v16bf b = load_b_tile(xT + (size_t)n0 * C + k0, C, lane);
    acc = wmma_bf16(a, b, acc);
  }
  int N = lane & 15, hh = lane >> 4;
  int col = n0 + N;                          // k = h*64 + w
  int i = ((col & 63) << 6) | (col >> 6);    // permuted: i = w*64 + h
#pragma unroll
  for (int r = 0; r < 8; ++r) {
    int m = m0 + r + (hh << 3);
    yT[(size_t)i * C8 + m] = (__bf16)(acc[r] + bpos[m]);
  }
}

// ---------------------------------------------------------------------------
// Gram pass 1: per-row online-softmax stats of S = yT @ yT^T  (K=64)
// ---------------------------------------------------------------------------
__global__ void __launch_bounds__(32)
k_gram_stats(const __bf16* __restrict__ yT, float* __restrict__ mbuf,
             float* __restrict__ zbuf) {
  int lane = threadIdx.x;
  int i0 = blockIdx.x << 4;
  const __bf16* A = yT + (size_t)i0 * C8;
  v16bf a0 = load_a_tile(A, C8, lane);
  v16bf a1 = load_a_tile(A + 32, C8, lane);
  float mr[8], zr[8];
#pragma unroll
  for (int r = 0; r < 8; ++r) { mr[r] = -3.0e38f; zr[r] = 0.f; }
  for (int jt = 0; jt < 256; ++jt) {
    const __bf16* Bp = yT + (size_t)jt * 16 * C8;
    v8f acc = {};
    acc = wmma_bf16(a0, load_b_tile(Bp, C8, lane), acc);
    acc = wmma_bf16(a1, load_b_tile(Bp + 32, C8, lane), acc);
#pragma unroll
    for (int r = 0; r < 8; ++r) {
      float s = acc[r];
      float nm = fmaxf(mr[r], s);
      zr[r] = zr[r] * __expf(mr[r] - nm) + __expf(s - nm);
      mr[r] = nm;
    }
  }
  // reduce over the 16 lanes of each half-wave (each half covers N=0..15)
#pragma unroll
  for (int off = 1; off < 16; off <<= 1) {
#pragma unroll
    for (int r = 0; r < 8; ++r) {
      float om = __shfl_xor(mr[r], off);
      float oz = __shfl_xor(zr[r], off);
      float nm = fmaxf(mr[r], om);
      zr[r] = zr[r] * __expf(mr[r] - nm) + oz * __expf(om - nm);
      mr[r] = nm;
    }
  }
  if ((lane & 15) == 0) {
    int base = i0 + ((lane >> 4) << 3);
#pragma unroll
    for (int r = 0; r < 8; ++r) {
      mbuf[base + r] = mr[r];
      zbuf[base + r] = 1.f / zr[r];
    }
  }
}

// ---------------------------------------------------------------------------
// Gram pass 2: X4T[j][i] = exp(S[j][i]-m_i)/Z_i  (valid by symmetry of S)
// ---------------------------------------------------------------------------
__global__ void __launch_bounds__(32)
k_x4t(const __bf16* __restrict__ yT, const float* __restrict__ mbuf,
      const float* __restrict__ zbuf, __bf16* __restrict__ X4T) {
  int lane = threadIdx.x;
  int j0 = blockIdx.y << 4;
  int icBase = blockIdx.x << 4;
  const __bf16* A = yT + (size_t)j0 * C8;
  v16bf a0 = load_a_tile(A, C8, lane);
  v16bf a1 = load_a_tile(A + 32, C8, lane);
  int N = lane & 15, hh = lane >> 4;
  for (int it = 0; it < 16; ++it) {
    int i0 = (icBase + it) << 4;
    const __bf16* Bp = yT + (size_t)i0 * C8;
    v8f acc = {};
    acc = wmma_bf16(a0, load_b_tile(Bp, C8, lane), acc);
    acc = wmma_bf16(a1, load_b_tile(Bp + 32, C8, lane), acc);
    int i = i0 + N;
    float mi = mbuf[i], rz = zbuf[i];
#pragma unroll
    for (int r = 0; r < 8; ++r) {
      int j = j0 + r + (hh << 3);
      X4T[(size_t)j * HW + i] = (__bf16)(__expf(acc[r] - mi) * rz);
    }
  }
}

// ---------------------------------------------------------------------------
// Big GEMM: x6 = x (512x4096) @ x4 (4096x4096), B columns read as X4T rows.
// 2x2 register blocking: each wave computes a 32x32 output block.
// ---------------------------------------------------------------------------
__global__ void __launch_bounds__(32)
k_gemm_x6(const __bf16* __restrict__ xb, const __bf16* __restrict__ X4T,
          float* __restrict__ x6) {
  int lane = threadIdx.x;
  int j0 = blockIdx.x << 5;   // 128 blocks
  int c0 = blockIdx.y << 5;   // 16 blocks
  const __bf16* A0 = xb + (size_t)c0 * HW;
  const __bf16* A1 = A0 + (size_t)16 * HW;
  const __bf16* B0 = X4T + (size_t)j0 * HW;
  const __bf16* B1 = B0 + (size_t)16 * HW;
  v8f acc00 = {}, acc01 = {}, acc10 = {}, acc11 = {};
  for (int k0 = 0; k0 < HW; k0 += 32) {
    v16bf a0 = load_a_tile(A0 + k0, HW, lane);
    v16bf a1 = load_a_tile(A1 + k0, HW, lane);
    v16bf b0 = load_b_tile(B0 + k0, HW, lane);
    v16bf b1 = load_b_tile(B1 + k0, HW, lane);
    acc00 = wmma_bf16(a0, b0, acc00);
    acc01 = wmma_bf16(a0, b1, acc01);
    acc10 = wmma_bf16(a1, b0, acc10);
    acc11 = wmma_bf16(a1, b1, acc11);
  }
  int N = lane & 15, hh = lane >> 4;
#pragma unroll
  for (int r = 0; r < 8; ++r) {
    size_t rowA = (size_t)(c0 + r + (hh << 3)) * HW;
    size_t rowB = rowA + (size_t)16 * HW;
    x6[rowA + j0 + N]      = acc00[r];
    x6[rowA + j0 + 16 + N] = acc01[r];
    x6[rowB + j0 + N]      = acc10[r];
    x6[rowB + j0 + 16 + N] = acc11[r];
  }
}

// ---------------------------------------------------------------------------
// Row softmax of x6, add x, softmax over each 64-wide W group,
// store x1_1 transposed as bf16 (HW x C) for the conv implicit GEMM.
// ---------------------------------------------------------------------------
__global__ void __launch_bounds__(256)
k_softmax_rows(const float* __restrict__ x6, const float* __restrict__ x,
               __bf16* __restrict__ x11T) {
  __shared__ float red[8];
  __shared__ float u[HW];
  int c = blockIdx.x;
  int t = threadIdx.x;
  const float* row = x6 + (size_t)c * HW;
  float lm = -3.0e38f;
  for (int j = t; j < HW; j += 256) lm = fmaxf(lm, row[j]);
#pragma unroll
  for (int o = 16; o > 0; o >>= 1) lm = fmaxf(lm, __shfl_xor(lm, o));
  if ((t & 31) == 0) red[t >> 5] = lm;
  __syncthreads();
  float m = red[0];
#pragma unroll
  for (int w = 1; w < 8; ++w) m = fmaxf(m, red[w]);
  __syncthreads();
  float ls = 0.f;
  for (int j = t; j < HW; j += 256) ls += __expf(row[j] - m);
#pragma unroll
  for (int o = 16; o > 0; o >>= 1) ls += __shfl_xor(ls, o);
  if ((t & 31) == 0) red[t >> 5] = ls;
  __syncthreads();
  float Z = 0.f;
#pragma unroll
  for (int w = 0; w < 8; ++w) Z += red[w];
  float rZ = 1.f / Z;
  for (int j = t; j < HW; j += 256)
    u[j] = __expf(row[j] - m) * rZ + x[(size_t)c * HW + j];
  __syncthreads();
  // per-h softmax over w: 64 groups of 64, 4 threads per group
  int g = t >> 2, q = t & 3;
  const float* gp = u + g * 64 + q * 16;
  float gm = -3.0e38f;
#pragma unroll
  for (int e = 0; e < 16; ++e) gm = fmaxf(gm, gp[e]);
  gm = fmaxf(gm, __shfl_xor(gm, 1));
  gm = fmaxf(gm, __shfl_xor(gm, 2));
  float gs = 0.f;
#pragma unroll
  for (int e = 0; e < 16; ++e) gs += __expf(gp[e] - gm);
  gs += __shfl_xor(gs, 1);
  gs += __shfl_xor(gs, 2);
  float gi = 1.f / gs;
#pragma unroll
  for (int e = 0; e < 16; ++e) {
    int j = g * 64 + q * 16 + e;
    x11T[(size_t)j * C + c] = (__bf16)(__expf(u[j] - gm) * gi);
  }
}

// ---------------------------------------------------------------------------
// 3x3 stride-2 pad-1 conv as implicit GEMM. inT is (HW, C) bf16.
// 2x2 blocking: 32 co x 32 output pixels per wave; K = 9 taps x 512 ci.
// Pad handled by zeroed B chunks (loads predicated; WMMA runs full-wave).
// ---------------------------------------------------------------------------
__global__ void __launch_bounds__(32)
k_conv3s2(const __bf16* __restrict__ inT, const __bf16* __restrict__ w3b,
          const float* __restrict__ b3, float* __restrict__ out) {
  int lane = threadIdx.x;
  int p0 = blockIdx.x << 5;    // 32 pixels
  int co0 = blockIdx.y << 5;   // 32 output channels
  int N = lane & 15, hh = lane >> 4;
  int pA = p0 + N, pB = p0 + 16 + N;
  v8f acc00 = {}, acc01 = {}, acc10 = {}, acc11 = {};
  vf4 zero = {0.f, 0.f, 0.f, 0.f};
  for (int kk = 0; kk < 9; ++kk) {
    int kh = kk / 3, kw = kk % 3;
    int ih0 = 2 * (pA >> 5) - 1 + kh, iw0 = 2 * (pA & 31) - 1 + kw;
    int ih1 = 2 * (pB >> 5) - 1 + kh, iw1 = 2 * (pB & 31) - 1 + kw;
    bool v0 = ((unsigned)ih0 < 64u) && ((unsigned)iw0 < 64u);
    bool v1 = ((unsigned)ih1 < 64u) && ((unsigned)iw1 < 64u);
    const __bf16* bb0 = inT + (size_t)(v0 ? ((ih0 << 6) + iw0) : 0) * C + (hh << 4);
    const __bf16* bb1 = inT + (size_t)(v1 ? ((ih1 << 6) + iw1) : 0) * C + (hh << 4);
    const __bf16* ab0 = w3b + (size_t)kk * C * C + (size_t)co0 * C;
    const __bf16* ab1 = ab0 + (size_t)16 * C;
    for (int k0 = 0; k0 < C; k0 += 32) {
      v16bf a0 = load_a_tile(ab0 + k0, C, lane);
      v16bf a1 = load_a_tile(ab1 + k0, C, lane);
      BfPack u0, u1;
      if (v0) {
        u0.q[0] = *(const vf4*)(bb0 + k0);
        u0.q[1] = *(const vf4*)(bb0 + k0 + 8);
      } else { u0.q[0] = zero; u0.q[1] = zero; }
      if (v1) {
        u1.q[0] = *(const vf4*)(bb1 + k0);
        u1.q[1] = *(const vf4*)(bb1 + k0 + 8);
      } else { u1.q[0] = zero; u1.q[1] = zero; }
      acc00 = wmma_bf16(a0, u0.v, acc00);
      acc01 = wmma_bf16(a0, u1.v, acc01);
      acc10 = wmma_bf16(a1, u0.v, acc10);
      acc11 = wmma_bf16(a1, u1.v, acc11);
    }
  }
#pragma unroll
  for (int r = 0; r < 8; ++r) {
    int coA = co0 + r + (hh << 3);
    int coB = coA + 16;
    float bA = b3[coA], bB = b3[coB];
    out[(size_t)coA * PO + pA] = acc00[r] + bA;
    out[(size_t)coA * PO + pB] = acc01[r] + bA;
    out[(size_t)coB * PO + pA] = acc10[r] + bB;
    out[(size_t)coB * PO + pB] = acc11[r] + bB;
  }
}

// ---------------------------------------------------------------------------
// Final per-channel stage: three 32x32 (K=32) WMMA matmuls with softmaxes,
// fused activation, bilinear 2x upsample to d_out.
// ---------------------------------------------------------------------------
DEVINL void mm32(const __bf16* P, const __bf16* Q, float* S, int lane,
                 int m0, int n0) {
  v16bf a = load_a_tile(P + m0 * 32, 32, lane);
  v16bf b = load_b_tile(Q + n0 * 32, 32, lane);
  v8f acc = {};
  acc = wmma_bf16(a, b, acc);
  int N = lane & 15, hh = lane >> 4;
#pragma unroll
  for (int r = 0; r < 8; ++r)
    S[(m0 + r + (hh << 3)) * 32 + n0 + N] = acc[r];
}

DEVINL void rowstats8(const float* r8, float& m, float& s) {
  m = -3.0e38f;
#pragma unroll
  for (int e = 0; e < 8; ++e) m = fmaxf(m, r8[e]);
  m = fmaxf(m, __shfl_xor(m, 1));
  m = fmaxf(m, __shfl_xor(m, 2));
  s = 0.f;
#pragma unroll
  for (int e = 0; e < 8; ++e) s += __expf(r8[e] - m);
  s += __shfl_xor(s, 1);
  s += __shfl_xor(s, 2);
}

__global__ void __launch_bounds__(128)
k_final(const float* __restrict__ x12, const float* __restrict__ c3,
        float* __restrict__ out) {
  __shared__ __align__(16) float  c3f[1024];
  __shared__ __align__(16) float  sbuf[1024];
  __shared__ __align__(16) float  x22[1024];
  __shared__ __align__(16) float  xf[1024];
  __shared__ __align__(16) __bf16 a1b[1024];
  __shared__ __align__(16) __bf16 c3b[1024];
  __shared__ __align__(16) __bf16 x43b[1024];
  __shared__ __align__(16) __bf16 x32b[1024];
  int c = blockIdx.x, t = threadIdx.x;
  const float* A1 = x12 + (size_t)c * 1024;
  const float* C3 = c3 + (size_t)c * 1024;
  for (int e = t; e < 1024; e += 128) {
    float a = A1[e], v = C3[e];
    c3f[e] = v;
    a1b[e] = (__bf16)a;
    c3b[e] = (__bf16)v;
    float l = v >= 0.f ? v : 0.2f * v;               // LeakyReLU(0.2)
    x43b[e] = (__bf16)(1.f / (1.f + __expf(-l)));    // sigmoid
  }
  __syncthreads();
  int lane = t & 31, wave = t >> 5;
  int m0 = (wave >> 1) << 4, n0 = (wave & 1) << 4;
  int row = t >> 2, q = t & 3;

  mm32(a1b, c3b, sbuf, lane, m0, n0);                // S1 = x1_2 @ c3^T
  __syncthreads();
  {
    const float* r8 = sbuf + row * 32 + q * 8;
    float m, s; rowstats8(r8, m, s);
    float inv = 1.f / s;
    float* d = x22 + row * 32 + q * 8;
#pragma unroll
    for (int e = 0; e < 8; ++e) d[e] = __expf(r8[e] - m) * inv;
  }
  __syncthreads();
  mm32(c3b, x43b, sbuf, lane, m0, n0);               // S2 = c3 @ x4_3^T
  __syncthreads();
  {
    const float* r8 = sbuf + row * 32 + q * 8;
    float m, s; rowstats8(r8, m, s);
    float inv = 1.f / s;
    __bf16* d = x32b + row * 32 + q * 8;
#pragma unroll
    for (int e = 0; e < 8; ++e) d[e] = (__bf16)(__expf(r8[e] - m) * inv);
  }
  __syncthreads();
  mm32(a1b, x32b, sbuf, lane, m0, n0);               // S3 = x1_2 @ x3_2^T
  __syncthreads();
  {
    int base = row * 32 + q * 8;
    const float* r8 = sbuf + base;
    float m, s; rowstats8(r8, m, s);
    float inv = 1.f / s;
#pragma unroll
    for (int e = 0; e < 8; ++e) {
      float v = __expf(r8[e] - m) * inv + x22[base + e] + c3f[base + e];
      xf[base + e] = v > 0.f ? v : 0.f;              // ReLU
    }
  }
  __syncthreads();
  // bilinear 2x upsample (half-pixel centers), 32x32 -> 64x64
  for (int e = t; e < 4096; e += 128) {
    int oy = e >> 6, ox = e & 63;
    float sy = oy * 0.5f - 0.25f;
    float sx = ox * 0.5f - 0.25f;
    int y0 = (int)floorf(sy), x0 = (int)floorf(sx);
    float fy = sy - (float)y0, fx = sx - (float)x0;
    int y0c = min(max(y0, 0), 31), y1c = min(max(y0 + 1, 0), 31);
    int x0c = min(max(x0, 0), 31), x1c = min(max(x0 + 1, 0), 31);
    float v00 = xf[y0c * 32 + x0c], v01 = xf[y0c * 32 + x1c];
    float v10 = xf[y1c * 32 + x0c], v11 = xf[y1c * 32 + x1c];
    float v = (1.f - fy) * ((1.f - fx) * v00 + fx * v01) +
              fy * ((1.f - fx) * v10 + fx * v11);
    out[(size_t)c * 4096 + e] = v;
  }
}

// ---------------------------------------------------------------------------
extern "C" void kernel_launch(void* const* d_in, const int* in_sizes, int n_in,
                              void* d_out, int out_size, void* d_ws, size_t ws_size,
                              hipStream_t stream) {
  const float* x    = (const float*)d_in[0];  // (1,512,64,64)
  const float* wpos = (const float*)d_in[1];  // (64,512,1,1)
  const float* bpos = (const float*)d_in[2];  // (64,)
  const float* w3   = (const float*)d_in[3];  // (512,512,3,3)
  const float* b3   = (const float*)d_in[4];  // (512,)
  float* out = (float*)d_out;

  char* ws = (char*)d_ws;
  __bf16* xb   = (__bf16*)(ws + OFF_XB);
  __bf16* xT   = (__bf16*)(ws + OFF_XT);
  __bf16* wb   = (__bf16*)(ws + OFF_WB);
  __bf16* w3b  = (__bf16*)(ws + OFF_W3B);
  __bf16* yT   = (__bf16*)(ws + OFF_YT);
  float*  mbuf = (float*)(ws + OFF_M);
  float*  zbuf = (float*)(ws + OFF_Z);
  __bf16* X4T  = (__bf16*)(ws + OFF_X4T);
  float*  x6   = (float*)(ws + OFF_X6);
  __bf16* x11T = (__bf16*)(ws + OFF_X11);
  float*  x12  = (float*)(ws + OFF_X12);
  float*  c3b_ = (float*)(ws + OFF_C3);

  k_pack_x   <<<(C * HW + 255) / 256, 256, 0, stream>>>(x, xb, xT);
  k_pack_wpos<<<(C8 * C + 255) / 256, 256, 0, stream>>>(wpos, wb);
  k_pack_w3  <<<(C * C * 9 + 255) / 256, 256, 0, stream>>>(w3, w3b);

  k_conv1x1   <<<dim3(256, 4),  32, 0, stream>>>(wb, xT, bpos, yT);
  k_gram_stats<<<256,           32, 0, stream>>>(yT, mbuf, zbuf);
  k_x4t       <<<dim3(16, 256), 32, 0, stream>>>(yT, mbuf, zbuf, X4T);
  k_gemm_x6   <<<dim3(128, 16), 32, 0, stream>>>(xb, X4T, x6);
  k_softmax_rows<<<512, 256, 0, stream>>>(x6, x, x11T);

  k_conv3s2<<<dim3(32, 16), 32, 0, stream>>>(xT,   w3b, b3, c3b_);  // c3
  k_conv3s2<<<dim3(32, 16), 32, 0, stream>>>(x11T, w3b, b3, x12);   // x1_2

  k_final<<<C, 128, 0, stream>>>(x12, c3b_, out);
}
